// DiffusionCell_56143812493778
// MI455X (gfx1250) — compile-verified, matrix operand order checked by reading, not run on hardware
//
#include <hip/hip_runtime.h>
#include <hip/hip_bf16.h>
#include <math.h>

// ---------------- problem constants ----------------
#define Hh       256
#define NNODES   16384
#define NEDGES   131072
#define BGR      16
#define MAXNODE  1024
#define NHEADS   8
#define HDIM     32
#define EPSc     1e-5f
#define SCALEc   0.17677669529663689f   // 1/sqrt(32)

typedef __attribute__((ext_vector_type(16))) __bf16 v16bf;
typedef __attribute__((ext_vector_type(8)))  float  v8f;

// f32 -> bf16 round-to-nearest-even
static __device__ __forceinline__ unsigned short f2bf(float x){
  union { float f; unsigned u; } c; c.f = x;
  unsigned u = c.u;
  u += 0x7fffu + ((u >> 16) & 1u);
  return (unsigned short)(u >> 16);
}
// two f32 -> packed bf16x2
static __device__ __forceinline__ unsigned pack2(float lo, float hi){
  return (unsigned)f2bf(lo) | ((unsigned)f2bf(hi) << 16);
}

union Frag16 { v16bf v; uint4 q[2]; };

// Load a 16-half WMMA fragment: 8 contiguous halves at p, 8 more at p+16
// (CDNA5 16-bit A/B layout: lanes<16 hold K {0..7,16..23}, lanes>=16 hold K {8..15,24..31})
static __device__ __forceinline__ v16bf load_frag(const unsigned short* p){
  Frag16 f;
  f.q[0] = *reinterpret_cast<const uint4*>(p);
  f.q[1] = *reinterpret_cast<const uint4*>(p + 16);
  return f.v;
}

static __device__ __forceinline__ v8f wmma_bf16(v16bf a, v16bf b, v8f c){
  return __builtin_amdgcn_wmma_f32_16x16x32_bf16(false, a, false, b, (short)0, c, false, false);
}

// ---------------- shared GEMM skeleton ----------------
// block = 256 threads = 8 wave32; macro tile 64 rows x 256 cols; per-wave 16x128 (8 accum tiles)
// Double-buffered LDS: As 2x[64][40] bf16 (k-tile of A), Bs 2x[256][40] bf16 ([n][k] transposed)
#define ASZ (64*40)
#define BSZ (256*40)

// vectorized A k-tile stage: thread -> 8 contiguous K elems of one row
static __device__ __forceinline__ void stageA(const float* A, long lda, long m0, int k0,
                                              unsigned short* As){
  int r  = threadIdx.x >> 2;
  int kq = (threadIdx.x & 3) * 8;
  const float* src = A + (m0 + r)*lda + k0 + kq;
  float4 f0 = *reinterpret_cast<const float4*>(src);
  float4 f1 = *reinterpret_cast<const float4*>(src + 4);
  uint4 pk;
  pk.x = pack2(f0.x, f0.y); pk.y = pack2(f0.z, f0.w);
  pk.z = pack2(f1.x, f1.y); pk.w = pack2(f1.z, f1.w);
  *reinterpret_cast<uint4*>(As + r*40 + kq) = pk;
}
// vectorized B k-tile stage (transposed): thread owns column n, coalesced global reads,
// packs 8 K values per ds_store_b128
static __device__ __forceinline__ void stageB(const float* B, long ldb, int k0, int n0,
                                              unsigned short* Bs){
  int n = threadIdx.x;
  const float* src = B + (long)k0*ldb + n0 + n;
#pragma unroll
  for (int kq = 0; kq < 4; ++kq){
    float f[8];
#pragma unroll
    for (int j = 0; j < 8; ++j) f[j] = src[(long)(kq*8 + j)*ldb];
    uint4 pk;
    pk.x = pack2(f[0], f[1]); pk.y = pack2(f[2], f[3]);
    pk.z = pack2(f[4], f[5]); pk.w = pack2(f[6], f[7]);
    *reinterpret_cast<uint4*>(Bs + n*40 + kq*8) = pk;
  }
}

static __device__ __forceinline__ void gemm_loop(const float* A, long lda, long m0,
                                                 const float* B, long ldb, int n0, int K,
                                                 unsigned short* As, unsigned short* Bs,
                                                 v8f acc[8], int wm, int wn, int lane){
  stageA(A, lda, m0, 0, As);
  stageB(B, ldb, 0, n0, Bs);
  __syncthreads();
  int p = 0;
  for (int k0 = 0; k0 < K; k0 += 32){
    int pn = p ^ 1;
    if (k0 + 32 < K){                       // stage next tile while computing current
      stageA(A, lda, m0, k0 + 32, As + pn*ASZ);
      stageB(B, ldb, k0 + 32, n0, Bs + pn*BSZ);
    }
    if (k0 + 64 < K)                        // global_prefetch_b8 for tile after next
      __builtin_prefetch(&B[(long)(k0 + 64)*ldb + n0 + (int)threadIdx.x], 0, 0);
    const unsigned short* Asp = As + p*ASZ;
    const unsigned short* Bsp = Bs + p*BSZ;
    v16bf a = load_frag(Asp + (16*wm + (lane & 15))*40 + 8*(lane >> 4));
    v16bf bf[8];
#pragma unroll
    for (int t = 0; t < 8; ++t)
      bf[t] = load_frag(Bsp + (128*wn + 16*t + (lane & 15))*40 + 8*(lane >> 4));
#pragma unroll
    for (int t = 0; t < 8; ++t)
      acc[t] = wmma_bf16(a, bf[t], acc[t]);
    __syncthreads();
    p = pn;
  }
}

static __device__ __forceinline__ void init_bias(v8f acc[8], const float* bias, int n0,
                                                 int wn, int lane){
#pragma unroll
  for (int t = 0; t < 8; ++t){
    float bv = bias[n0 + 128*wn + 16*t + (lane & 15)];
    v8f a;
#pragma unroll
    for (int r = 0; r < 8; ++r) a[r] = bv;
    acc[t] = a;
  }
}

// C/D layout: col = n0 + 128*wn + 16*t + (lane&15) ; row = m0 + 16*wm + 8*(lane>>4) + r

// ---------------- kernels ----------------

__global__ void k_zero(float* p, size_t n){
  size_t i = (size_t)blockIdx.x*blockDim.x + threadIdx.x;
  size_t st = (size_t)gridDim.x*blockDim.x;
  for (; i < n; i += st) p[i] = 0.f;
}

// h_e_inj = [h_e | rho[src] | rho[dst] | f] @ W_ei + b_ei  (K=256 WMMA + 3 rank-1 terms)
__global__ __launch_bounds__(256) void k_inject_edges(const float* h_e, const float* rho,
    const float* fsk, const int* eidx, const float* W_ei, const float* b_ei, float* out){
  __shared__ __align__(16) unsigned short As[2*ASZ];
  __shared__ __align__(16) unsigned short Bs[2*BSZ];
  const int lane = threadIdx.x & 31, w = threadIdx.x >> 5, wm = w & 3, wn = w >> 2;
  const long m0 = (long)blockIdx.x * 64;
  v8f acc[8]; init_bias(acc, b_ei, 0, wn, lane);
  gemm_loop(h_e, Hh, m0, W_ei, Hh, 0, Hh, As, Bs, acc, wm, wn, lane);
  float rs[8], rd[8], fe[8];
#pragma unroll
  for (int r = 0; r < 8; ++r){
    long e = m0 + 16*wm + 8*(lane >> 4) + r;
    int se = eidx[e], de = eidx[NEDGES + e];
    rs[r] = rho[se]; rd[r] = rho[de]; fe[r] = fsk[e];
  }
#pragma unroll
  for (int t = 0; t < 8; ++t){
    int col = 128*wn + 16*t + (lane & 15);
    float w1 = W_ei[256*Hh + col], w2 = W_ei[257*Hh + col], w3 = W_ei[258*Hh + col];
#pragma unroll
    for (int r = 0; r < 8; ++r){
      long row = m0 + 16*wm + 8*(lane >> 4) + r;
      out[row*Hh + col] = acc[t][r] + rs[r]*w1 + rd[r]*w2 + fe[r]*w3;
    }
  }
}

// h_v_inj = [h_v | rho] @ W_ni + b_ni
__global__ __launch_bounds__(256) void k_inject_nodes(const float* h_v, const float* rho,
    const float* W_ni, const float* b_ni, float* out){
  __shared__ __align__(16) unsigned short As[2*ASZ];
  __shared__ __align__(16) unsigned short Bs[2*BSZ];
  const int lane = threadIdx.x & 31, w = threadIdx.x >> 5, wm = w & 3, wn = w >> 2;
  const long m0 = (long)blockIdx.x * 64;
  v8f acc[8]; init_bias(acc, b_ni, 0, wn, lane);
  gemm_loop(h_v, Hh, m0, W_ni, Hh, 0, Hh, As, Bs, acc, wm, wn, lane);
  float rr[8];
#pragma unroll
  for (int r = 0; r < 8; ++r) rr[r] = rho[m0 + 16*wm + 8*(lane >> 4) + r];
#pragma unroll
  for (int t = 0; t < 8; ++t){
    int col = 128*wn + 16*t + (lane & 15);
    float wz = W_ni[256*Hh + col];
#pragma unroll
    for (int r = 0; r < 8; ++r){
      long row = m0 + 16*wm + 8*(lane >> 4) + r;
      out[row*Hh + col] = acc[t][r] + rr[r]*wz;
    }
  }
}

// 4 node linears sharing the same A operand (A, B, D, E)
__global__ __launch_bounds__(256) void k_lin4(const float* X,
    const float* W0, const float* B0, float* O0,
    const float* W1, const float* B1, float* O1,
    const float* W2, const float* B2, float* O2,
    const float* W3, const float* B3, float* O3){
  __shared__ __align__(16) unsigned short As[2*ASZ];
  __shared__ __align__(16) unsigned short Bs[2*BSZ];
  const int lane = threadIdx.x & 31, w = threadIdx.x >> 5, wm = w & 3, wn = w >> 2;
  const long m0 = (long)blockIdx.x * 64;
  const float* Ws[4] = {W0, W1, W2, W3};
  const float* Bb[4] = {B0, B1, B2, B3};
  float*       Os[4] = {O0, O1, O2, O3};
  for (int s = 0; s < 4; ++s){
    v8f acc[8]; init_bias(acc, Bb[s], 0, wn, lane);
    gemm_loop(X, Hh, m0, Ws[s], Hh, 0, Hh, As, Bs, acc, wm, wn, lane);
    float* O = Os[s];
#pragma unroll
    for (int t = 0; t < 8; ++t){
      int col = 128*wn + 16*t + (lane & 15);
#pragma unroll
      for (int r = 0; r < 8; ++r){
        long row = m0 + 16*wm + 8*(lane >> 4) + r;
        O[row*Hh + col] = acc[t][r];
      }
    }
  }
}

// plain [M,256]@[256,256]+bias (used for Ce -> e_msg)
__global__ __launch_bounds__(256) void k_gemm_bias(const float* X, const float* W,
    const float* bias, float* out){
  __shared__ __align__(16) unsigned short As[2*ASZ];
  __shared__ __align__(16) unsigned short Bs[2*BSZ];
  const int lane = threadIdx.x & 31, w = threadIdx.x >> 5, wm = w & 3, wn = w >> 2;
  const long m0 = (long)blockIdx.x * 64;
  v8f acc[8]; init_bias(acc, bias, 0, wn, lane);
  gemm_loop(X, Hh, m0, W, Hh, 0, Hh, As, Bs, acc, wm, wn, lane);
#pragma unroll
  for (int t = 0; t < 8; ++t){
    int col = 128*wn + 16*t + (lane & 15);
#pragma unroll
    for (int r = 0; r < 8; ++r){
      long row = m0 + 16*wm + 8*(lane >> 4) + r;
      out[row*Hh + col] = acc[t][r];
    }
  }
}

// e_msg += Dx[dst] + Ex[src]; sigma scatter into num/den; per-channel BN partials
__global__ __launch_bounds__(256) void k_edge_msg(const float* Dx, const float* Ex,
    const float* Bx, const int* eidx, float* emsg, float* num, float* den, float* part){
  const long e0 = (long)blockIdx.x * 64;
  const int c = threadIdx.x;
  float s1 = 0.f, s2 = 0.f;
  for (int i = 0; i < 64; ++i){
    long e = e0 + i;
    int se = eidx[e], de = eidx[NEDGES + e];
    long ie = e*Hh + c;
    float m = emsg[ie] + Dx[(long)de*Hh + c] + Ex[(long)se*Hh + c];
    emsg[ie] = m;
    float sg = 1.f / (1.f + __expf(-m));
    atomicAdd(&num[(long)de*Hh + c], sg * Bx[(long)se*Hh + c]);
    atomicAdd(&den[(long)de*Hh + c], sg);
    s1 += m; s2 += m*m;
  }
  part[(size_t)blockIdx.x*512 + c]       = s1;
  part[(size_t)blockIdx.x*512 + 256 + c] = s2;
}

// deterministic per-channel mean / rsqrt(var+eps) from block partials
__global__ void k_finalize(const float* part, int nblk, float invN, float* mu, float* inv){
  int c = threadIdx.x;
  float s1 = 0.f, s2 = 0.f;
  for (int b = 0; b < nblk; ++b){
    s1 += part[(size_t)b*512 + c];
    s2 += part[(size_t)b*512 + 256 + c];
  }
  float m = s1*invN;
  mu[c]  = m;
  inv[c] = rsqrtf(s2*invN - m*m + EPSc);
}

// h_e_new = h_e_inj + relu(bn_e(e_msg))
__global__ void k_he_new(const float* hei, const float* emsg, const float* mu, const float* inv,
                         const float* g, const float* bb, float* out){
  long i = (long)blockIdx.x*256 + threadIdx.x;
  int c = threadIdx.x;
  float bn = g[c]*(emsg[i]-mu[c])*inv[c] + bb[c];
  out[i] = hei[i] + fmaxf(bn, 0.f);
}

// x_pre = Ax + num/(den+1e-6) (in place over Ax) with BN partials
__global__ __launch_bounds__(256) void k_xpre(const float* num, const float* den,
                                              float* ax, float* part){
  const long n0 = (long)blockIdx.x * 64;
  const int c = threadIdx.x;
  float s1 = 0.f, s2 = 0.f;
  for (int i = 0; i < 64; ++i){
    long idx = (n0 + i)*Hh + c;
    float v = ax[idx] + num[idx] / (den[idx] + 1e-6f);
    ax[idx] = v;
    s1 += v; s2 += v*v;
  }
  part[(size_t)blockIdx.x*512 + c]       = s1;
  part[(size_t)blockIdx.x*512 + 256 + c] = s2;
}

// x = h_v_inj + relu(bn_x(x_pre))
__global__ void k_x(const float* hvi, const float* xpre, const float* mu, const float* inv,
                    const float* g, const float* bb, float* out){
  long i = (long)blockIdx.x*256 + threadIdx.x;
  int c = threadIdx.x;
  float bn = g[c]*(xpre[i]-mu[c])*inv[c] + bb[c];
  out[i] = hvi[i] + fmaxf(bn, 0.f);
}

// QKV projection: 3 n-chunks of W_qkv [256,768]
__global__ __launch_bounds__(256) void k_qkv(const float* X, const float* Wq, const float* bq,
                                             float* Q, float* Kb, float* V){
  __shared__ __align__(16) unsigned short As[2*ASZ];
  __shared__ __align__(16) unsigned short Bs[2*BSZ];
  const int lane = threadIdx.x & 31, w = threadIdx.x >> 5, wm = w & 3, wn = w >> 2;
  const long m0 = (long)blockIdx.x * 64;
  float* outs[3] = {Q, Kb, V};
  for (int ci = 0; ci < 3; ++ci){
    v8f acc[8]; init_bias(acc, bq, 256*ci, wn, lane);
    gemm_loop(X, Hh, m0, Wq, 3*Hh, 256*ci, Hh, As, Bs, acc, wm, wn, lane);
    float* O = outs[ci];
#pragma unroll
    for (int t = 0; t < 8; ++t){
      int col = 128*wn + 16*t + (lane & 15);
#pragma unroll
      for (int r = 0; r < 8; ++r){
        long row = m0 + 16*wm + 8*(lane >> 4) + r;
        O[row*Hh + col] = acc[t][r];
      }
    }
  }
}

// flash attention: grid (qchunk=8, head=8, graph=16); 128 queries/block, K/V chunks of 128
__global__ __launch_bounds__(256) void k_attn(const float* Q, const float* Kt,
                                              const float* V, float* O){
  __shared__ __align__(16) unsigned short Qs[128*40];
  __shared__ __align__(16) unsigned short KVs[128*40];       // K chunk, reused as V^T [32][136]
  __shared__ __align__(16) unsigned short Pls[8*16*136];     // per-wave P (A-layout)
  const int lane = threadIdx.x & 31, w = threadIdx.x >> 5;
  const int qc = blockIdx.x, h = blockIdx.y, b = blockIdx.z;
  const long nodeQ0 = (long)b*MAXNODE + (long)qc*128;
  const long nodeK0 = (long)b*MAXNODE;
  const int colh = h*HDIM;
  {  // vectorized Q stage: thread -> 16 contiguous dims of one row
    int r = threadIdx.x >> 1, dq = (threadIdx.x & 1)*16;
    const float* src = Q + (nodeQ0 + r)*Hh + colh + dq;
    float4 f0 = *reinterpret_cast<const float4*>(src);
    float4 f1 = *reinterpret_cast<const float4*>(src + 4);
    float4 f2 = *reinterpret_cast<const float4*>(src + 8);
    float4 f3 = *reinterpret_cast<const float4*>(src + 12);
    uint4 p0, p1;
    p0.x = pack2(f0.x,f0.y); p0.y = pack2(f0.z,f0.w);
    p0.z = pack2(f1.x,f1.y); p0.w = pack2(f1.z,f1.w);
    p1.x = pack2(f2.x,f2.y); p1.y = pack2(f2.z,f2.w);
    p1.z = pack2(f3.x,f3.y); p1.w = pack2(f3.z,f3.w);
    *reinterpret_cast<uint4*>(Qs + r*40 + dq)     = p0;
    *reinterpret_cast<uint4*>(Qs + r*40 + dq + 8) = p1;
  }
  __syncthreads();
  v16bf qf = load_frag(Qs + (16*w + (lane & 15))*40 + 8*(lane >> 4));
  float mrow[8], lrow[8];
  v8f o0, o1;
#pragma unroll
  for (int r = 0; r < 8; ++r){ mrow[r] = -1e30f; lrow[r] = 0.f; o0[r] = 0.f; o1[r] = 0.f; }
  unsigned short* pw = Pls + w*16*136;
  for (int kc = 0; kc < MAXNODE/128; ++kc){
    __syncthreads();
    {  // vectorized K stage
      int r = threadIdx.x >> 1, dq = (threadIdx.x & 1)*16;
      const float* src = Kt + (nodeK0 + kc*128 + r)*Hh + colh + dq;
      float4 f0 = *reinterpret_cast<const float4*>(src);
      float4 f1 = *reinterpret_cast<const float4*>(src + 4);
      float4 f2 = *reinterpret_cast<const float4*>(src + 8);
      float4 f3 = *reinterpret_cast<const float4*>(src + 12);
      uint4 p0, p1;
      p0.x = pack2(f0.x,f0.y); p0.y = pack2(f0.z,f0.w);
      p0.z = pack2(f1.x,f1.y); p0.w = pack2(f1.z,f1.w);
      p1.x = pack2(f2.x,f2.y); p1.y = pack2(f2.z,f2.w);
      p1.z = pack2(f3.x,f3.y); p1.w = pack2(f3.z,f3.w);
      *reinterpret_cast<uint4*>(KVs + r*40 + dq)     = p0;
      *reinterpret_cast<uint4*>(KVs + r*40 + dq + 8) = p1;
    }
    __syncthreads();
    v8f zc;
#pragma unroll
    for (int r = 0; r < 8; ++r) zc[r] = 0.f;
    v8f s[8];
#pragma unroll
    for (int j = 0; j < 8; ++j){
      v16bf bk = load_frag(KVs + (16*j + (lane & 15))*40 + 8*(lane >> 4));
      s[j] = wmma_bf16(qf, bk, zc);
    }
    float cm[8];
#pragma unroll
    for (int r = 0; r < 8; ++r) cm[r] = -1e30f;
#pragma unroll
    for (int j = 0; j < 8; ++j)
#pragma unroll
      for (int r = 0; r < 8; ++r){ float v = s[j][r]*SCALEc; s[j][r] = v; cm[r] = fmaxf(cm[r], v); }
#pragma unroll
    for (int msk = 1; msk < 16; msk <<= 1)
#pragma unroll
      for (int r = 0; r < 8; ++r) cm[r] = fmaxf(cm[r], __shfl_xor(cm[r], msk, 32));
    float corr[8], csum[8];
#pragma unroll
    for (int r = 0; r < 8; ++r){
      float mn = fmaxf(mrow[r], cm[r]);
      corr[r] = __expf(mrow[r] - mn);
      mrow[r] = mn; csum[r] = 0.f;
    }
#pragma unroll
    for (int j = 0; j < 8; ++j)
#pragma unroll
      for (int r = 0; r < 8; ++r){ float p = __expf(s[j][r] - mrow[r]); s[j][r] = p; csum[r] += p; }
#pragma unroll
    for (int msk = 1; msk < 16; msk <<= 1)
#pragma unroll
      for (int r = 0; r < 8; ++r) csum[r] += __shfl_xor(csum[r], msk, 32);
#pragma unroll
    for (int r = 0; r < 8; ++r){
      lrow[r] = lrow[r]*corr[r] + csum[r];
      o0[r] *= corr[r]; o1[r] *= corr[r];
    }
#pragma unroll
    for (int j = 0; j < 8; ++j)
#pragma unroll
      for (int r = 0; r < 8; ++r)
        pw[(r + 8*(lane >> 4))*136 + 16*j + (lane & 15)] = f2bf(s[j][r]);
    __syncthreads();
    {  // vectorized V^T stage over K chunk: thread owns dim d, 16 contiguous keys
      int d = threadIdx.x & 31, rq = (threadIdx.x >> 5)*16;
      float f[16];
#pragma unroll
      for (int j = 0; j < 16; ++j)
        f[j] = V[(nodeK0 + kc*128 + rq + j)*Hh + colh + d];
      uint4 p0, p1;
      p0.x = pack2(f[0],f[1]);   p0.y = pack2(f[2],f[3]);
      p0.z = pack2(f[4],f[5]);   p0.w = pack2(f[6],f[7]);
      p1.x = pack2(f[8],f[9]);   p1.y = pack2(f[10],f[11]);
      p1.z = pack2(f[12],f[13]); p1.w = pack2(f[14],f[15]);
      *reinterpret_cast<uint4*>(KVs + d*136 + rq)     = p0;
      *reinterpret_cast<uint4*>(KVs + d*136 + rq + 8) = p1;
    }
    __syncthreads();
#pragma unroll
    for (int kk = 0; kk < 4; ++kk){
      v16bf ap  = load_frag(pw + (lane & 15)*136 + 32*kk + 8*(lane >> 4));
      v16bf bv0 = load_frag(KVs + (lane & 15)*136 + 32*kk + 8*(lane >> 4));
      o0 = wmma_bf16(ap, bv0, o0);
      v16bf bv1 = load_frag(KVs + (16 + (lane & 15))*136 + 32*kk + 8*(lane >> 4));
      o1 = wmma_bf16(ap, bv1, o1);
    }
  }
#pragma unroll
  for (int r = 0; r < 8; ++r){
    long row = nodeQ0 + 16*w + 8*(lane >> 4) + r;
    O[row*Hh + colh + (lane & 15)]      = o0[r] / lrow[r];
    O[row*Hh + colh + 16 + (lane & 15)] = o1[r] / lrow[r];
  }
}

// GEMM + bias + residual + row LayerNorm (used for O-proj/LN1 and FFN2/LN2)
__global__ __launch_bounds__(256) void k_gemm_ln(const float* X, long lda, int K,
    const float* W, const float* bias, const float* resid,
    const float* g, const float* bvec, float* out){
  __shared__ __align__(16) unsigned short As[2*ASZ];
  __shared__ __align__(16) unsigned short Bs[2*BSZ];
  __shared__ float s_sum[64][2], s_sq[64][2], s_mu[64], s_inv[64];
  const int lane = threadIdx.x & 31, w = threadIdx.x >> 5, wm = w & 3, wn = w >> 2;
  const long m0 = (long)blockIdx.x * 64;
  v8f acc[8]; init_bias(acc, bias, 0, wn, lane);
  gemm_loop(X, lda, m0, W, Hh, 0, K, As, Bs, acc, wm, wn, lane);
  float p1[8], p2[8];
#pragma unroll
  for (int r = 0; r < 8; ++r){ p1[r] = 0.f; p2[r] = 0.f; }
#pragma unroll
  for (int t = 0; t < 8; ++t){
    int col = 128*wn + 16*t + (lane & 15);
#pragma unroll
    for (int r = 0; r < 8; ++r){
      long row = m0 + 16*wm + 8*(lane >> 4) + r;
      float v = acc[t][r] + resid[row*Hh + col];
      acc[t][r] = v; p1[r] += v; p2[r] += v*v;
    }
  }
#pragma unroll
  for (int msk = 1; msk < 16; msk <<= 1)
#pragma unroll
    for (int r = 0; r < 8; ++r){
      p1[r] += __shfl_xor(p1[r], msk, 32);
      p2[r] += __shfl_xor(p2[r], msk, 32);
    }
  if ((lane & 15) == 0){
#pragma unroll
    for (int r = 0; r < 8; ++r){
      int row = 16*wm + 8*(lane >> 4) + r;
      s_sum[row][wn] = p1[r]; s_sq[row][wn] = p2[r];
    }
  }
  __syncthreads();
  if (threadIdx.x < 64){
    float S1 = s_sum[threadIdx.x][0] + s_sum[threadIdx.x][1];
    float S2 = s_sq[threadIdx.x][0] + s_sq[threadIdx.x][1];
    float m = S1 / 256.f;
    s_mu[threadIdx.x]  = m;
    s_inv[threadIdx.x] = rsqrtf(S2/256.f - m*m + EPSc);
  }
  __syncthreads();
#pragma unroll
  for (int t = 0; t < 8; ++t){
    int col = 128*wn + 16*t + (lane & 15);
    float gc = g[col], bc = bvec[col];
#pragma unroll
    for (int r = 0; r < 8; ++r){
      int rl = 16*wm + 8*(lane >> 4) + r;
      out[(m0 + rl)*Hh + col] = gc*(acc[t][r] - s_mu[rl])*s_inv[rl] + bc;
    }
  }
}

// FFN1: relu(X @ W_f1 + b_f1), 4 n-chunks -> [N,1024]
__global__ __launch_bounds__(256) void k_ffn1(const float* X, const float* W, const float* bias,
                                              float* O){
  __shared__ __align__(16) unsigned short As[2*ASZ];
  __shared__ __align__(16) unsigned short Bs[2*BSZ];
  const int lane = threadIdx.x & 31, w = threadIdx.x >> 5, wm = w & 3, wn = w >> 2;
  const long m0 = (long)blockIdx.x * 64;
  for (int ci = 0; ci < 4; ++ci){
    v8f acc[8]; init_bias(acc, bias, 256*ci, wn, lane);
    gemm_loop(X, Hh, m0, W, 4*Hh, 256*ci, Hh, As, Bs, acc, wm, wn, lane);
#pragma unroll
    for (int t = 0; t < 8; ++t){
      int col = 128*wn + 16*t + (lane & 15);
#pragma unroll
      for (int r = 0; r < 8; ++r){
        long row = m0 + 16*wm + 8*(lane >> 4) + r;
        O[row*(4*Hh) + 256*ci + col] = fmaxf(acc[t][r], 0.f);
      }
    }
  }
}

// delta_f = relu(h_e_new @ W_r1 + b_r1) @ W_r2 + b_r2  (fused row-dot, no 134MB intermediate)
__global__ __launch_bounds__(256) void k_delta(const float* HE, const float* W1, const float* b1,
    const float* w2, const float* b2, float* dout){
  __shared__ __align__(16) unsigned short As[2*ASZ];
  __shared__ __align__(16) unsigned short Bs[2*BSZ];
  __shared__ float s_d[64][2];
  const int lane = threadIdx.x & 31, w = threadIdx.x >> 5, wm = w & 3, wn = w >> 2;
  const long m0 = (long)blockIdx.x * 64;
  v8f acc[8]; init_bias(acc, b1, 0, wn, lane);
  gemm_loop(HE, Hh, m0, W1, Hh, 0, Hh, As, Bs, acc, wm, wn, lane);
  float p[8];
#pragma unroll
  for (int r = 0; r < 8; ++r) p[r] = 0.f;
#pragma unroll
  for (int t = 0; t < 8; ++t){
    int col = 128*wn + 16*t + (lane & 15);
    float wc = w2[col];
#pragma unroll
    for (int r = 0; r < 8; ++r) p[r] += fmaxf(acc[t][r], 0.f)*wc;
  }
#pragma unroll
  for (int msk = 1; msk < 16; msk <<= 1)
#pragma unroll
    for (int r = 0; r < 8; ++r) p[r] += __shfl_xor(p[r], msk, 32);
  if ((lane & 15) == 0){
#pragma unroll
    for (int r = 0; r < 8; ++r) s_d[16*wm + 8*(lane >> 4) + r][wn] = p[r];
  }
  __syncthreads();
  if (threadIdx.x < 64)
    dout[m0 + threadIdx.x] = s_d[threadIdx.x][0] + s_d[threadIdx.x][1] + b2[0];
}

// ---------------- launcher ----------------
extern "C" void kernel_launch(void* const* d_in, const int* in_sizes, int n_in,
                              void* d_out, int out_size, void* d_ws, size_t ws_size,
                              hipStream_t stream) {
  const float* h_v   = (const float*)d_in[0];
  const float* h_e   = (const float*)d_in[1];
  const float* rho_v = (const float*)d_in[2];
  const float* fsk   = (const float*)d_in[3];
  const int*   eidx  = (const int*)  d_in[4];
  // d_in[5] = batch_vec (unused: equal-sized graphs -> reshape)
  const float* W_ei  = (const float*)d_in[6];  const float* b_ei  = (const float*)d_in[7];
  const float* W_ni  = (const float*)d_in[8];  const float* b_ni  = (const float*)d_in[9];
  const float* W_A   = (const float*)d_in[10]; const float* b_A   = (const float*)d_in[11];
  const float* W_B   = (const float*)d_in[12]; const float* b_B   = (const float*)d_in[13];
  const float* W_C   = (const float*)d_in[14]; const float* b_C   = (const float*)d_in[15];
  const float* W_D   = (const float*)d_in[16]; const float* b_D   = (const float*)d_in[17];
  const float* W_E   = (const float*)d_in[18]; const float* b_E   = (const float*)d_in[19];
  const float* bnxg  = (const float*)d_in[20]; const float* bnxb  = (const float*)d_in[21];
  const float* bneg  = (const float*)d_in[22]; const float* bneb  = (const float*)d_in[23];
  const float* ln1g  = (const float*)d_in[24]; const float* ln1b  = (const float*)d_in[25];
  const float* ln2g  = (const float*)d_in[26]; const float* ln2b  = (const float*)d_in[27];
  const float* W_qkv = (const float*)d_in[28]; const float* b_qkv = (const float*)d_in[29];
  const float* W_o   = (const float*)d_in[30]; const float* b_o   = (const float*)d_in[31];
  const float* W_f1  = (const float*)d_in[32]; const float* b_f1  = (const float*)d_in[33];
  const float* W_f2  = (const float*)d_in[34]; const float* b_f2  = (const float*)d_in[35];
  const float* W_r1  = (const float*)d_in[36]; const float* b_r1  = (const float*)d_in[37];
  const float* W_r2  = (const float*)d_in[38]; const float* b_r2  = (const float*)d_in[39];

  const size_t NH = (size_t)NNODES*Hh;   // 4,194,304
  const size_t EH = (size_t)NEDGES*Hh;   // 33,554,432
  float* ws   = (float*)d_ws;
  float* HVI  = ws;                      // h_v_inj               [NH]
  float* HEI  = HVI + NH;                // h_e_inj, later H1     [EH]
  float* AX   = HEI + EH;                // Ax, later x_pre       [NH]
  float* BX   = AX  + NH;                // Bx, later Q           [NH]
  float* DX   = BX  + NH;                // Dx, later K           [NH]
  float* EX   = DX  + NH;                // Ex, later V           [NH]
  float* EMSG = EX  + NH;                // Ce / e_msg            [EH]
  float* NUM  = EMSG + EH;               // num, later attn O     [NH]
  float* DEN  = NUM + NH;                // den, later xd2        [NH]
  float* XB   = DEN + NH;                // x (post-BN residual)  [NH]
  float* PART = XB  + NH;                // BN partials           [2048*512]
  float* STAT = PART + (size_t)2048*512; // mu_e, inv_e, mu_x, inv_x

  float* out_hv = (float*)d_out;         // h_v_new [NH]
  float* out_he = out_hv + NH;           // h_e_new [EH]
  float* out_df = out_he + EH;           // delta_f [NEDGES]

  const int gE = NEDGES/64, gN = NNODES/64;

  k_zero<<<2048, 256, 0, stream>>>(NUM, 2*NH);                         // num, den
  k_inject_edges<<<gE, 256, 0, stream>>>(h_e, rho_v, fsk, eidx, W_ei, b_ei, HEI);
  k_inject_nodes<<<gN, 256, 0, stream>>>(h_v, rho_v, W_ni, b_ni, HVI);
  k_lin4<<<gN, 256, 0, stream>>>(HVI, W_A, b_A, AX, W_B, b_B, BX,
                                 W_D, b_D, DX, W_E, b_E, EX);
  k_gemm_bias<<<gE, 256, 0, stream>>>(HEI, W_C, b_C, EMSG);            // Ce
  k_edge_msg<<<gE, 256, 0, stream>>>(DX, EX, BX, eidx, EMSG, NUM, DEN, PART);
  k_finalize<<<1, 256, 0, stream>>>(PART, gE, 1.f/NEDGES, STAT, STAT + 256);
  k_he_new<<<(int)(EH/256), 256, 0, stream>>>(HEI, EMSG, STAT, STAT + 256, bneg, bneb, out_he);
  k_delta<<<gE, 256, 0, stream>>>(out_he, W_r1, b_r1, W_r2, b_r2, out_df);
  k_xpre<<<gN, 256, 0, stream>>>(NUM, DEN, AX, PART);
  k_finalize<<<1, 256, 0, stream>>>(PART, gN, 1.f/NNODES, STAT + 512, STAT + 768);
  k_x<<<(int)(NH/256), 256, 0, stream>>>(HVI, AX, STAT + 512, STAT + 768, bnxg, bnxb, XB);
  k_qkv<<<gN, 256, 0, stream>>>(XB, W_qkv, b_qkv, BX, DX, EX);         // Q,K,V reuse Bx,Dx,Ex
  k_attn<<<dim3(MAXNODE/128, NHEADS, BGR), 256, 0, stream>>>(BX, DX, EX, NUM);
  k_gemm_ln<<<gN, 256, 0, stream>>>(NUM, Hh, Hh, W_o, b_o, XB, ln1g, ln1b, DEN);   // xd2
  k_ffn1<<<gN, 256, 0, stream>>>(DEN, W_f1, b_f1, HEI);                // H1 reuses HEI
  k_gemm_ln<<<gN, 256, 0, stream>>>(HEI, 4*Hh, 4*Hh, W_f2, b_f2, DEN, ln2g, ln2b, out_hv);
}
